// GraphTransformerEncoderLayer_15882789060652
// MI455X (gfx1250) — matrix operand order repaired and checked by reading, not compiled, not run on hardware
//
#include <hip/hip_runtime.h>
#include <hip/hip_bf16.h>

// ---------------------------------------------------------------------------
// GraphTransformerEncoderLayer for MI455X (gfx1250), f16 WMMA + fp32 VALU.
// B=2, N=16384, K=32, D_IN=D_H=128, H=4, DH=32
// ---------------------------------------------------------------------------

#define BD 2
#define ND 16384
#define KN 32          // neighbors
#define DM 128         // model dim
#define NH 4           // heads
#define DH 32          // head dim
#define TILE 16        // nodes per workgroup
#define EPSV 1e-6f

typedef __attribute__((ext_vector_type(16))) _Float16 v16h;
typedef __attribute__((ext_vector_type(8)))  _Float16 v8h;
typedef __attribute__((ext_vector_type(4)))  _Float16 v4h;
typedef __attribute__((ext_vector_type(8)))  float    v8f;
typedef __attribute__((ext_vector_type(4)))  float    v4f;

__device__ __forceinline__ v8f wmma16(v16h a, v16h b, v8f c) {
  return __builtin_amdgcn_wmma_f32_16x16x32_f16(false, a, false, b, (short)0, c,
                                                false, false);
}

__device__ __forceinline__ v4f ntload4(const v4f* p) {
  return __builtin_nontemporal_load(p);
}

// A-fragment (16x32 f16, M x K) from f16 LDS tile (row-major, strideH halves,
// rows 16B aligned). Per ISA 7.12.2: lane L holds row M=L%16; halves 0..7 are
// K = kc*32 + (L>>4)*8 + 0..7 ; halves 8..15 are that +16.
__device__ __forceinline__ v16h load_afrag(const _Float16* base, int strideH,
                                           int mt, int kc, int lane) {
  const _Float16* p =
      base + (mt * 16 + (lane & 15)) * strideH + kc * 32 + ((lane >> 4) << 3);
  v8h lo = *(const v8h*)(p);
  v8h hi = *(const v8h*)(p + 16);
  return __builtin_shufflevector(lo, hi, 0, 1, 2, 3, 4, 5, 6, 7, 8, 9, 10, 11,
                                 12, 13, 14, 15);
}

// B-fragment from pre-packed weights: fragment (kc,nt) is 1024B, lane-major.
__device__ __forceinline__ v16h load_bfrag(const _Float16* pack, int ntc,
                                           int kc, int nt, int lane) {
  return *(const v16h*)(pack + ((size_t)(kc * ntc + nt) * 512) + lane * 16);
}

// ---------------------------------------------------------------------------
// Pack fp32 weight [Kd x Nd] (row-major) into f16 WMMA B-fragment order.
// out[f*512 + L*16 + j] = W[(kc*32 + (L>>4)*8 + (j&7) + (j>>3)*16) * Nd + nt*16 + (L&15)]
// ---------------------------------------------------------------------------
__global__ __launch_bounds__(256) void pack_w_kernel(const float* __restrict__ W,
                                                     _Float16* __restrict__ out,
                                                     int ntc, int Nd) {
  int tid = blockIdx.x * 256 + threadIdx.x;
  int j = tid & 15;
  int L = (tid >> 4) & 31;
  int f = tid >> 9;
  int nt = f % ntc;
  int kc = f / ntc;
  int Kk = kc * 32 + ((L >> 4) << 3) + (j & 7) + ((j >> 3) << 4);
  int Nn = nt * 16 + (L & 15);
  out[tid] = (_Float16)W[(size_t)Kk * Nd + Nn];
}

// ---------------------------------------------------------------------------
// Main fused kernel: one 16-node tile per workgroup, 256 threads (8 waves).
// ---------------------------------------------------------------------------
__global__ __launch_bounds__(256) void gte_layer_kernel(
    const float* __restrict__ h_V, const float* __restrict__ h_E,
    const float* __restrict__ mask, const _Float16* __restrict__ wQp,
    const _Float16* __restrict__ wKp, const _Float16* __restrict__ wVp,
    const _Float16* __restrict__ wOp, const _Float16* __restrict__ wInP,
    const _Float16* __restrict__ wOutP, const float* __restrict__ b_in,
    const float* __restrict__ b_out, const float* __restrict__ g0,
    const float* __restrict__ bn0, const float* __restrict__ g1,
    const float* __restrict__ bn1, float* __restrict__ out) {
  __shared__ _Float16 eT[32 * 136];    // E_n tile, f16
  __shared__ float    kpT[32 * 129];   // Kp f32
  __shared__ float    vpT[32 * 129];   // Vp f32
  __shared__ _Float16 hvH[16 * 136];   // h_V tile f16 (A for Q)
  __shared__ float    hvF[16 * 128];   // h_V tile f32 (residual)
  __shared__ float    qT[16 * 128];    // Q tile f32
  __shared__ _Float16 updH[16 * 136];  // attention update, f16
  __shared__ float    dhT[16 * 128];   // upd @ W_O, f32
  __shared__ float    hF[16 * 128];    // post-LN0 f32
  __shared__ _Float16 hH[16 * 136];    // post-LN0 f16 (A for FFN1)
  __shared__ _Float16 actH[16 * 520];  // relu(h@W_in+b) f16 (A for FFN2)
  __shared__ float    ffT[16 * 128];   // FFN out f32

  const int tid = threadIdx.x;
  const int wave = tid >> 5;
  const int lane = tid & 31;
  const int fn0 = blockIdx.x * TILE;  // flat node index (b*N+n), tile start

  // E staging coordinates (8 threads/row, 16 cols each) + register pipeline.
  const int er = tid >> 3;
  const int ec = (tid & 7) * 16;
  v4f ebuf[4];
  {
    const v4f* s = (const v4f*)&h_E[((size_t)fn0 * KN + er) * DM + ec];
#pragma unroll
    for (int q = 0; q < 4; ++q) ebuf[q] = ntload4(&s[q]);
  }

  // ---- Stage 0: load h_V tile (f32 + f16 copies) -------------------------
  {
    int row = tid >> 4;
    int c = (tid & 15) * 8;
    const v4f* s = (const v4f*)&h_V[(size_t)(fn0 + row) * DM + c];
    v4f a = ntload4(&s[0]), b = ntload4(&s[1]);
    float* fd = &hvF[row * DM + c];
    fd[0] = a.x; fd[1] = a.y; fd[2] = a.z; fd[3] = a.w;
    fd[4] = b.x; fd[5] = b.y; fd[6] = b.z; fd[7] = b.w;
    v4h p0 = {(_Float16)a.x, (_Float16)a.y, (_Float16)a.z, (_Float16)a.w};
    v4h p1 = {(_Float16)b.x, (_Float16)b.y, (_Float16)b.z, (_Float16)b.w};
    *(v4h*)&hvH[row * 136 + c] = p0;
    *(v4h*)&hvH[row * 136 + c + 4] = p1;
  }

  // Persistent B fragments for K/V projection (held in VGPRs all kernel).
  const _Float16* kvp = (wave >> 2) ? wVp : wKp;
  const int nt0 = (wave & 3) * 2;
  v16h bkv[4][2];
#pragma unroll
  for (int kc = 0; kc < 4; ++kc) {
#pragma unroll
    for (int t = 0; t < 2; ++t)
      bkv[kc][t] = load_bfrag(kvp, 8, kc, nt0 + t, lane);
  }

  __syncthreads();

  // ---- Q = h_V @ W_Q (one 16x16 tile per wave) ---------------------------
  {
    v8f acc = {};
#pragma unroll
    for (int kc = 0; kc < 4; ++kc) {
      v16h a = load_afrag(hvH, 136, 0, kc, lane);
      acc = wmma16(a, load_bfrag(wQp, 8, kc, wave, lane), acc);
    }
#pragma unroll
    for (int r = 0; r < 8; ++r)
      qT[((lane >> 4) * 8 + r) * DM + wave * 16 + (lane & 15)] = acc[r];
  }
  __syncthreads();

  // ---- Node loop: K/V projection + attention -----------------------------
  for (int i = 0; i < TILE; ++i) {
    const int fn = fn0 + i;
    // Commit pipelined E_n registers -> f16 LDS.
    {
#pragma unroll
      for (int q = 0; q < 4; ++q) {
        v4f v = ebuf[q];
        v4h p = {(_Float16)v.x, (_Float16)v.y, (_Float16)v.z, (_Float16)v.w};
        *(v4h*)&eT[er * 136 + ec + q * 4] = p;
      }
    }
    __syncthreads();

    // Kick off node i+1's E loads; they retire at next iteration's commit.
    if (i + 1 < TILE) {
      const v4f* s = (const v4f*)&h_E[((size_t)(fn + 1) * KN + er) * DM + ec];
#pragma unroll
      for (int q = 0; q < 4; ++q) ebuf[q] = ntload4(&s[q]);
    }

    // Kp / Vp: wave -> (matrix = wave>>2, N-tiles nt0..nt0+1, both M-tiles)
    {
      v8f a00 = {}, a01 = {}, a10 = {}, a11 = {};
#pragma unroll
      for (int kc = 0; kc < 4; ++kc) {
        v16h a0 = load_afrag(eT, 136, 0, kc, lane);
        v16h a1 = load_afrag(eT, 136, 1, kc, lane);
        a00 = wmma16(a0, bkv[kc][0], a00);
        a01 = wmma16(a0, bkv[kc][1], a01);
        a10 = wmma16(a1, bkv[kc][0], a10);
        a11 = wmma16(a1, bkv[kc][1], a11);
      }
      float* dst = (wave >> 2) ? vpT : kpT;
      int nc = nt0 * 16 + (lane & 15);
#pragma unroll
      for (int r = 0; r < 8; ++r) {
        int m0 = (lane >> 4) * 8 + r;
        dst[m0 * 129 + nc] = a00[r];
        dst[m0 * 129 + nc + 16] = a01[r];
        dst[(16 + m0) * 129 + nc] = a10[r];
        dst[(16 + m0) * 129 + nc + 16] = a11[r];
      }
    }
    __syncthreads();

    // Attention: waves 0..3, one head each; lane = neighbor k (then dim d).
    if (wave < 4) {
      const int h = wave;
      const int k = lane;
      float mk = __builtin_nontemporal_load(&mask[(size_t)fn * KN + k]);
      const float* qrow = &qT[i * DM + h * DH];
      const float* krow = &kpT[k * 129 + h * DH];
      float s = 0.f;
#pragma unroll
      for (int d = 0; d < DH; ++d) s += qrow[d] * krow[d];
      float lg = (mk > 0.f) ? s * 0.17677669529663687f : -3.0e38f;
      float mx = lg;
#pragma unroll
      for (int off = 16; off >= 1; off >>= 1)
        mx = fmaxf(mx, __shfl_xor(mx, off, 32));
      float e = __expf(lg - mx);
      float sm = e;
#pragma unroll
      for (int off = 16; off >= 1; off >>= 1) sm += __shfl_xor(sm, off, 32);
      float att = (e / sm) * mk;
      // upd[h, d=lane] = sum_k att[k] * Vp[k, h*32+d]
      float u = 0.f;
#pragma unroll
      for (int k2 = 0; k2 < KN; ++k2) {
        float a = __shfl(att, k2, 32);
        u += a * vpT[k2 * 129 + h * DH + lane];
      }
      updH[i * 136 + h * DH + lane] = (_Float16)u;
    }
    __syncthreads();
  }

  // ---- dh = upd @ W_O ----------------------------------------------------
  {
    v8f acc = {};
#pragma unroll
    for (int kc = 0; kc < 4; ++kc) {
      v16h a = load_afrag(updH, 136, 0, kc, lane);
      acc = wmma16(a, load_bfrag(wOp, 8, kc, wave, lane), acc);
    }
#pragma unroll
    for (int r = 0; r < 8; ++r)
      dhT[((lane >> 4) * 8 + r) * DM + wave * 16 + (lane & 15)] = acc[r];
  }
  __syncthreads();

  // ---- LN0: h = LN(h_V + dh) --------------------------------------------
  {
    int row = tid >> 4;
    int cb = (tid & 15) * 8;
    float x[8], s = 0.f, s2 = 0.f;
#pragma unroll
    for (int j = 0; j < 8; ++j) {
      float v = hvF[row * DM + cb + j] + dhT[row * DM + cb + j];
      x[j] = v; s += v; s2 += v * v;
    }
#pragma unroll
    for (int off = 8; off >= 1; off >>= 1) {
      s += __shfl_xor(s, off, 32);
      s2 += __shfl_xor(s2, off, 32);
    }
    float mean = s * (1.f / DM);
    float var = s2 * (1.f / DM) - mean * mean;
    float inv = rsqrtf(var + EPSV);
#pragma unroll
    for (int j = 0; j < 8; ++j) {
      float y = (x[j] - mean) * inv * g0[cb + j] + bn0[cb + j];
      hF[row * DM + cb + j] = y;
      hH[row * 136 + cb + j] = (_Float16)y;
    }
  }
  __syncthreads();

  // ---- FFN1: act = relu(h @ W_in + b_in), [16x512] -----------------------
  {
    v16h aH[4];
#pragma unroll
    for (int kc = 0; kc < 4; ++kc) aH[kc] = load_afrag(hH, 136, 0, kc, lane);
#pragma unroll
    for (int t = 0; t < 4; ++t) {
      int nt = wave * 4 + t;
      v8f acc = {};
#pragma unroll
      for (int kc = 0; kc < 4; ++kc)
        acc = wmma16(aH[kc], load_bfrag(wInP, 32, kc, nt, lane), acc);
      int col = nt * 16 + (lane & 15);
      float bi = b_in[col];
#pragma unroll
      for (int r = 0; r < 8; ++r) {
        float v = acc[r] + bi;
        v = v > 0.f ? v : 0.f;
        actH[((lane >> 4) * 8 + r) * 520 + col] = (_Float16)v;
      }
    }
  }
  __syncthreads();

  // ---- FFN2: ff = act @ W_out + b_out, [16x128] --------------------------
  {
    v8f acc = {};
#pragma unroll
    for (int kc = 0; kc < 16; ++kc)
      acc = wmma16(load_afrag(actH, 520, 0, kc, lane),
                   load_bfrag(wOutP, 8, kc, wave, lane), acc);
    int col = wave * 16 + (lane & 15);
    float bo = b_out[col];
#pragma unroll
    for (int r = 0; r < 8; ++r)
      ffT[((lane >> 4) * 8 + r) * DM + col] = acc[r] + bo;
  }
  __syncthreads();

  // ---- LN1 and store (nontemporal: write-once stream) --------------------
  {
    int row = tid >> 4;
    int cb = (tid & 15) * 8;
    float x[8], s = 0.f, s2 = 0.f;
#pragma unroll
    for (int j = 0; j < 8; ++j) {
      float v = hF[row * DM + cb + j] + ffT[row * DM + cb + j];
      x[j] = v; s += v; s2 += v * v;
    }
#pragma unroll
    for (int off = 8; off >= 1; off >>= 1) {
      s += __shfl_xor(s, off, 32);
      s2 += __shfl_xor(s2, off, 32);
    }
    float mean = s * (1.f / DM);
    float var = s2 * (1.f / DM) - mean * mean;
    float inv = rsqrtf(var + EPSV);
#pragma unroll
    for (int j = 0; j < 8; ++j) {
      float y = (x[j] - mean) * inv * g1[cb + j] + bn1[cb + j];
      __builtin_nontemporal_store(y, &out[(size_t)(fn0 + row) * DM + cb + j]);
    }
  }
}

// ---------------------------------------------------------------------------
extern "C" void kernel_launch(void* const* d_in, const int* in_sizes, int n_in,
                              void* d_out, int out_size, void* d_ws,
                              size_t ws_size, hipStream_t stream) {
  const float* h_V = (const float*)d_in[0];
  const float* h_E = (const float*)d_in[1];
  const float* maskp = (const float*)d_in[2];
  const float* W_Q = (const float*)d_in[3];
  const float* W_K = (const float*)d_in[4];
  const float* W_V = (const float*)d_in[5];
  const float* W_O = (const float*)d_in[6];
  const float* W_in = (const float*)d_in[7];
  const float* b_in = (const float*)d_in[8];
  const float* W_out = (const float*)d_in[9];
  const float* b_out = (const float*)d_in[10];
  const float* g0 = (const float*)d_in[11];
  const float* bn0 = (const float*)d_in[12];
  const float* g1 = (const float*)d_in[13];
  const float* bn1 = (const float*)d_in[14];

  _Float16* ws = (_Float16*)d_ws;  // offsets in halves
  _Float16* wQp = ws;              // 4*8*512   = 16384 halves
  _Float16* wKp = ws + 16384;
  _Float16* wVp = ws + 32768;
  _Float16* wOp = ws + 49152;
  _Float16* wInP = ws + 65536;     // 4*32*512  = 65536 halves
  _Float16* wOutP = ws + 131072;   // 16*8*512  = 65536 halves

  pack_w_kernel<<<64, 256, 0, stream>>>(W_Q, wQp, 8, 128);
  pack_w_kernel<<<64, 256, 0, stream>>>(W_K, wKp, 8, 128);
  pack_w_kernel<<<64, 256, 0, stream>>>(W_V, wVp, 8, 128);
  pack_w_kernel<<<64, 256, 0, stream>>>(W_O, wOp, 8, 128);
  pack_w_kernel<<<256, 256, 0, stream>>>(W_in, wInP, 32, 512);
  pack_w_kernel<<<256, 256, 0, stream>>>(W_out, wOutP, 8, 128);

  const int tiles = (BD * ND) / TILE;  // 2048
  gte_layer_kernel<<<tiles, 256, 0, stream>>>(
      h_V, h_E, maskp, wQp, wKp, wVp, wOp, wInP, wOutP, b_in, b_out, g0, bn0,
      g1, bn1, (float*)d_out);
}
